// HeteroGraphSAGE_32839319945244
// MI455X (gfx1250) — matrix operand order repaired and checked by reading, not compiled, not run on hardware
//
#include <hip/hip_runtime.h>

// ---------------------------------------------------------------------------
// HeteroGraphSAGE on MI455X (gfx1250): f16 WMMA GEMMs + f32 atomic scatters.
// Each wave computes a full 16xN output strip (N=128/256) so the activation
// matrix A is streamed from HBM exactly once per GEMM.
// ---------------------------------------------------------------------------

typedef _Float16 f16;
typedef __attribute__((ext_vector_type(16))) _Float16 v16h;
typedef __attribute__((ext_vector_type(4)))  _Float16 v4h;
typedef __attribute__((ext_vector_type(8)))  float    v8f;
typedef __attribute__((ext_vector_type(4)))  float    v4f;

#define N_USERS  100000
#define N_ISSUES 200000
#define F_IN  128
#define F_HID 256

// ---- WMMA fragment loaders (contiguous k order: element s of lane l is k = 16*(l>>4) + s, same in A and B) ----------

// A-matrix 16x32 f16: lanes 0-15 = rows, K=0..15; lanes 16-31
// same rows, K=16..31. Source is f32 row-major [M][K].
__device__ __forceinline__ v16h load_a_frag(const float* __restrict__ A,
                                            int m0, int K, int kb, int lane) {
  const int m  = m0 + (lane & 15);
  const int kh = (lane >> 4) << 4;                 // 0 or 16
  const float* p = A + (size_t)m * K + (kb << 5);
  v4f a0 = *(const v4f*)(p + kh);
  v4f a1 = *(const v4f*)(p + kh + 4);
  v4f a2 = *(const v4f*)(p + kh + 8);
  v4f a3 = *(const v4f*)(p + kh + 12);
  v16h r;
#pragma unroll
  for (int i = 0; i < 4; ++i) {
    r[i]      = (f16)a0[i];
    r[4 + i]  = (f16)a1[i];
    r[8 + i]  = (f16)a2[i];
    r[12 + i] = (f16)a3[i];
  }
  return r;
}

// B-matrix 32x16 f16 (K x N). y = x @ W^T means B[k][n] = W[n][k], so column n
// is a contiguous row of W. Lanes 0-15: N=0..15, K=0..15; lanes 16-31: same N,
// K=16..31; K packed in ascending pairs. W is pre-converted f16 row-major [N][K].
__device__ __forceinline__ v16h load_b_frag(const f16* __restrict__ W,
                                            int n0, int K, int kb, int lane) {
  const int n  = n0 + (lane & 15);
  const int kh = (lane >> 4) << 4;                 // 0 or 16
  return *(const v16h*)(W + (size_t)n * K + (kb << 5) + kh);
}

// Fused Y = A1 @ W1^T [+ A2 @ W2^T] + bias, optional ReLU. NT = N/16 tiles,
// all held in accumulators so A is read once. 4 waves/block, one 16-row strip
// per wave.
template <int NT>
__global__ void __launch_bounds__(128)
gemm_fused(const float* __restrict__ A1, const f16* __restrict__ W1,
           const float* __restrict__ A2, const f16* __restrict__ W2,
           const float* __restrict__ bias, float* __restrict__ Y,
           int M, int K, int relu) {
  const int N = NT * 16;
  const int lane = threadIdx.x & 31;
  const int wave = threadIdx.x >> 5;
  const int m0 = (blockIdx.x * 4 + wave) << 4;
  if (m0 >= M) return;                               // wave-uniform: EXEC stays all-1s

  v8f acc[NT];
#pragma unroll
  for (int t = 0; t < NT; ++t) acc[t] = v8f{};

  const int ksteps = K >> 5;
  for (int kb = 0; kb < ksteps; ++kb) {
    v16h a = load_a_frag(A1, m0, K, kb, lane);
#pragma unroll
    for (int t = 0; t < NT; ++t) {
      v16h b = load_b_frag(W1, t * 16, K, kb, lane);
      acc[t] = __builtin_amdgcn_wmma_f32_16x16x32_f16(false, a, false, b,
                                                      (short)0, acc[t], false, false);
    }
  }
  if (A2) {
    for (int kb = 0; kb < ksteps; ++kb) {
      v16h a = load_a_frag(A2, m0, K, kb, lane);
#pragma unroll
      for (int t = 0; t < NT; ++t) {
        v16h b = load_b_frag(W2, t * 16, K, kb, lane);
        acc[t] = __builtin_amdgcn_wmma_f32_16x16x32_f16(false, a, false, b,
                                                        (short)0, acc[t], false, false);
      }
    }
  }

  // C/D layout: lane l -> col 16*t + (l&15); VGPR r -> row m0 + 8*(l>>4) + r.
  const int nl = lane & 15;
  const int mh = (lane >> 4) << 3;
#pragma unroll
  for (int t = 0; t < NT; ++t) {
    const int col = t * 16 + nl;
    const float bv = bias[col];
#pragma unroll
    for (int r = 0; r < 8; ++r) {
      float v = acc[t][r] + bv;
      if (relu) v = fmaxf(v, 0.0f);
      Y[(size_t)(m0 + mh + r) * N + col] = v;
    }
  }
}

// agg[dst[e]] += feat[src[e]]  (segment_sum), float4 gather + 4x f32 atomics.
__global__ void __launch_bounds__(256)
scatter_add(const float* __restrict__ feat, const int* __restrict__ src,
            const int* __restrict__ dst, float* __restrict__ agg,
            long long nwork, int fshift, int F) {
  long long t = (long long)blockIdx.x * blockDim.x + threadIdx.x;
  if (t >= nwork) return;
  const long long e = t >> fshift;
  const int c = ((int)(t & ((1LL << fshift) - 1))) << 2;
  const int s = src[e];
  const int d = dst[e];
  const v4f v = *(const v4f*)(feat + (size_t)s * F + c);
  float* q = agg + (size_t)d * F + c;
  unsafeAtomicAdd(q + 0, v[0]);
  unsafeAtomicAdd(q + 1, v[1]);
  unsafeAtomicAdd(q + 2, v[2]);
  unsafeAtomicAdd(q + 3, v[3]);
}

__global__ void __launch_bounds__(256)
zero_f4(float* __restrict__ p, long long n4) {
  long long t = (long long)blockIdx.x * blockDim.x + threadIdx.x;
  if (t < n4) ((v4f*)p)[t] = v4f{};
}

__global__ void __launch_bounds__(256)
cvt_f16v4(const float* __restrict__ s, f16* __restrict__ d, int n4) {
  int i = blockIdx.x * blockDim.x + threadIdx.x;
  if (i >= n4) return;
  v4f v = ((const v4f*)s)[i];
  v4h h;
  h[0] = (f16)v[0]; h[1] = (f16)v[1]; h[2] = (f16)v[2]; h[3] = (f16)v[3];
  ((v4h*)d)[i] = h;
}

extern "C" void kernel_launch(void* const* d_in, const int* in_sizes, int n_in,
                              void* d_out, int out_size, void* d_ws, size_t ws_size,
                              hipStream_t stream) {
  (void)n_in; (void)out_size; (void)ws_size;
  const float* x_user       = (const float*)d_in[0];
  const float* x_issue      = (const float*)d_in[1];
  const float* W_mlp        = (const float*)d_in[2];
  const float* b_mlp        = (const float*)d_in[3];
  const float* c1_ui_rel_W  = (const float*)d_in[4];
  const float* c1_ui_rel_b  = (const float*)d_in[5];
  const float* c1_ui_root_W = (const float*)d_in[6];
  const float* c1_iu_rel_W  = (const float*)d_in[7];
  const float* c1_iu_rel_b  = (const float*)d_in[8];
  const float* c1_iu_root_W = (const float*)d_in[9];
  const float* c2_ui_rel_W  = (const float*)d_in[10];
  const float* c2_ui_rel_b  = (const float*)d_in[11];
  const float* c2_ui_root_W = (const float*)d_in[12];
  const float* c2_iu_rel_W  = (const float*)d_in[13];
  const float* c2_iu_rel_b  = (const float*)d_in[14];
  const float* c2_iu_root_W = (const float*)d_in[15];
  const int*   src_user     = (const int*)d_in[16];
  const int*   dst_issue    = (const int*)d_in[17];
  const long long E = in_sizes[16];

  // ---- workspace carve-out (f32 activations, f16 weights at the tail) -----
  float* ws   = (float*)d_ws;
  size_t off  = 0;
  float* xi   = ws + off; off += (size_t)N_ISSUES * F_IN;   // MLP output
  float* aggI = ws + off; off += (size_t)N_ISSUES * F_HID;  // reused conv1/conv2
  float* aggU = ws + off; off += (size_t)N_USERS  * F_HID;  // reused conv1/conv2
  float* hI   = ws + off; off += (size_t)N_ISSUES * F_HID;
  float* hU   = ws + off; off += (size_t)N_USERS  * F_HID;
  f16* wb = (f16*)(ws + off);
  f16* h_mlp      = wb;               // 128*128
  f16* h_c1uirel  = h_mlp     + 16384;
  f16* h_c1uiroot = h_c1uirel + 32768;
  f16* h_c1iurel  = h_c1uiroot+ 32768;
  f16* h_c1iuroot = h_c1iurel + 32768;
  f16* h_c2uirel  = h_c1iuroot+ 32768;
  f16* h_c2uiroot = h_c2uirel + 32768;
  f16* h_c2iurel  = h_c2uiroot+ 32768;
  f16* h_c2iuroot = h_c2iurel + 32768;

  float* out_issue = (float*)d_out;
  float* out_user  = out_issue + (size_t)N_ISSUES * F_IN;

  auto cvt = [&](const float* s, f16* d, int n) {
    int n4 = n >> 2;
    cvt_f16v4<<<(n4 + 255) / 256, 256, 0, stream>>>(s, d, n4);
  };
  auto zero = [&](float* p, long long n) {
    long long n4 = n >> 2;
    zero_f4<<<(unsigned)((n4 + 255) / 256), 256, 0, stream>>>(p, n4);
  };
  auto gemm = [&](const float* A1, const f16* W1, const float* A2, const f16* W2,
                  const float* bias, float* Y, int M, int N, int K, int relu) {
    dim3 g((M / 16 + 3) / 4);
    if (N == 128)
      gemm_fused<8><<<g, 128, 0, stream>>>(A1, W1, A2, W2, bias, Y, M, K, relu);
    else
      gemm_fused<16><<<g, 128, 0, stream>>>(A1, W1, A2, W2, bias, Y, M, K, relu);
  };
  auto scat = [&](const float* feat, const int* s, const int* d, float* agg, int F) {
    int fshift = (F == 128) ? 5 : 6;          // F/4 float4-chunks per edge
    long long nw = E << fshift;
    scatter_add<<<(unsigned)((nw + 255) / 256), 256, 0, stream>>>(feat, s, d, agg, nw, fshift, F);
  };

  // ---- weight precision conversion ----------------------------------------
  cvt(W_mlp,        h_mlp,      128 * 128);
  cvt(c1_ui_rel_W,  h_c1uirel,  256 * 128);
  cvt(c1_ui_root_W, h_c1uiroot, 256 * 128);
  cvt(c1_iu_rel_W,  h_c1iurel,  256 * 128);
  cvt(c1_iu_root_W, h_c1iuroot, 256 * 128);
  cvt(c2_ui_rel_W,  h_c2uirel,  128 * 256);
  cvt(c2_ui_root_W, h_c2uiroot, 128 * 256);
  cvt(c2_iu_rel_W,  h_c2iurel,  128 * 256);
  cvt(c2_iu_root_W, h_c2iuroot, 128 * 256);

  // ---- issue MLP: xi = x_issue @ W_mlp^T + b ------------------------------
  gemm(x_issue, h_mlp, nullptr, nullptr, b_mlp, xi, N_ISSUES, F_IN, F_IN, 0);

  // ---- conv1 --------------------------------------------------------------
  zero(aggI, (long long)N_ISSUES * F_IN);
  zero(aggU, (long long)N_USERS  * F_IN);
  scat(x_user, src_user, dst_issue, aggI, F_IN);               // user -> issue
  gemm(aggI, h_c1uirel, xi, h_c1uiroot, c1_ui_rel_b, hI,
       N_ISSUES, F_HID, F_IN, 1);                              // relu fused
  scat(xi, dst_issue, src_user, aggU, F_IN);                   // issue -> user
  gemm(aggU, h_c1iurel, x_user, h_c1iuroot, c1_iu_rel_b, hU,
       N_USERS, F_HID, F_IN, 1);

  // ---- conv2 --------------------------------------------------------------
  zero(aggI, (long long)N_ISSUES * F_HID);
  zero(aggU, (long long)N_USERS  * F_HID);
  scat(hU, src_user, dst_issue, aggI, F_HID);
  gemm(aggI, h_c2uirel, hI, h_c2uiroot, c2_ui_rel_b, out_issue,
       N_ISSUES, F_IN, F_HID, 0);
  scat(hI, dst_issue, src_user, aggU, F_HID);
  gemm(aggU, h_c2iurel, hU, h_c2iuroot, c2_iu_rel_b, out_user,
       N_USERS, F_IN, F_HID, 0);
}
